// WITT_Encoder_3865470566406
// MI455X (gfx1250) — compile-verified, matrix operand order checked by reading, not run on hardware
//
#include <hip/hip_runtime.h>

// ---------------------------------------------------------------------------
// Types / WMMA helpers (CDNA5 gfx1250, wave32)
// ---------------------------------------------------------------------------
typedef __bf16 bf16;
typedef __attribute__((ext_vector_type(16))) __bf16 v16bf;
typedef __attribute__((ext_vector_type(8)))  __bf16 v8bf;
typedef __attribute__((ext_vector_type(8)))  float  v8f;
typedef __attribute__((ext_vector_type(4)))  float  f32x4;

#define WMMA_BF16(a, b, c) \
  __builtin_amdgcn_wmma_f32_16x16x32_bf16(false, (a), false, (b), (short)0, (c), false, false)

// Per CDNA5 ISA 16-bit A layout, lane needs row r = rbase + (lane&15),
// K chunks [kb, kb+8) and [16+kb, 16+kb+8) with kb = (lane&16) ? 8 : 0.
// With row-major LDS (ld bf16/row, ld % 8 == 0) both chunks are 16B-aligned
// contiguous runs -> two ds_load_b128 per fragment.
__device__ __forceinline__ v16bf frag_b128(const bf16* p, int ld, int rbase, int k0, int lane) {
  int r  = rbase + (lane & 15);
  int kb = (lane & 16) ? 8 : 0;
  const bf16* row = p + r * ld + k0 + kb;
  v8bf lo = *(const v8bf*)(row);
  v8bf hi = *(const v8bf*)(row + 16);
  return __builtin_shufflevector(lo, hi, 0, 1, 2, 3, 4, 5, 6, 7, 8, 9, 10, 11, 12, 13, 14, 15);
}

// ---------------------------------------------------------------------------
// Generic fused GEMM: C[M,N] = act(A[M,K] @ W[K,N] + bias) * rowscale (+/-resid)
// Block: 128 threads (4 wave32). C-tile 32(M) x 64(N); each wave owns one
// 16-row band and two 16-col tiles -> 2 WMMAs per k-step, A-frag reused.
// W is staged TRANSPOSED in LDS (Bt[n][k]) so B-fragments use the same
// contiguous b128 pattern as A-fragments.
// act: 0 none, 1 gelu(exact), 2 sigmoid, 3 relu
// resid_mode: 0 none, 1 out += resid, 2 out = resid * out (gating)
// ---------------------------------------------------------------------------
__global__ void gemm_kernel(const float* __restrict__ A, const float* __restrict__ W,
                            const float* __restrict__ bias, const float* __restrict__ resid,
                            const float* __restrict__ rowscale, float* __restrict__ C,
                            int M, int N, int K, int act, int resid_mode, int rpg) {
  __shared__ bf16 As[32][40];   // 32 rows x 32 K, stride 40 (80B, 16B aligned)
  __shared__ bf16 Bt[64][40];   // 64 cols x 32 K (transposed W tile)
  int tid   = threadIdx.x;
  int mBase = blockIdx.y * 32, nBase = blockIdx.x * 64;
  int wave  = tid >> 5, lane = tid & 31;
  int wm  = (wave & 1) * 16;
  int wn0 = (wave >> 1) * 16;          // second tile at wn0 + 32
  v8f acc0 = {0.f, 0.f, 0.f, 0.f, 0.f, 0.f, 0.f, 0.f};
  v8f acc1 = {0.f, 0.f, 0.f, 0.f, 0.f, 0.f, 0.f, 0.f};

  for (int kt = 0; kt < K; kt += 32) {
    bool fast = (mBase + 32 <= M) && (nBase + 64 <= N) && (kt + 32 <= K);
    if (fast) {
      // ---- A tile: thread -> (row, 8-float segment), float4 loads, packed store
      {
        int r = tid >> 2, seg = (tid & 3) * 8;
        const f32x4* pa = (const f32x4*)&A[(size_t)(mBase + r) * K + kt + seg];
        f32x4 a0 = pa[0], a1 = pa[1];
        v8bf pk;
        pk[0] = (bf16)a0[0]; pk[1] = (bf16)a0[1]; pk[2] = (bf16)a0[2]; pk[3] = (bf16)a0[3];
        pk[4] = (bf16)a1[0]; pk[5] = (bf16)a1[1]; pk[6] = (bf16)a1[2]; pk[7] = (bf16)a1[3];
        *(v8bf*)&As[r][seg] = pk;
      }
      // ---- W tile transposed: thread -> (col n, 16-K half); coalesced b32 loads
      {
        int n = tid & 63, kh = (tid >> 6) * 16;
        float v[16];
#pragma unroll
        for (int kk = 0; kk < 16; ++kk)
          v[kk] = W[(size_t)(kt + kh + kk) * N + nBase + n];
        v8bf p0, p1;
#pragma unroll
        for (int kk = 0; kk < 8; ++kk) { p0[kk] = (bf16)v[kk]; p1[kk] = (bf16)v[kk + 8]; }
        *(v8bf*)&Bt[n][kh]     = p0;
        *(v8bf*)&Bt[n][kh + 8] = p1;
      }
      if (kt + 32 < K) {  // speculative prefetch of next k-tile (global_prefetch_b8)
        __builtin_prefetch(&A[(size_t)(mBase + (tid >> 2)) * K + kt + 32], 0, 1);
        __builtin_prefetch(&W[(size_t)(kt + 32 + (tid >> 2)) * N + nBase + (tid & 3) * 16], 0, 1);
      }
    } else {
      // ---- guarded scalar staging (edge tiles), zero-fill OOB
      for (int idx = tid; idx < 1024; idx += 128) {
        int r = idx >> 5, cc = idx & 31;
        int gm = mBase + r, gk = kt + cc;
        As[r][cc] = (gm < M && gk < K) ? (bf16)A[(size_t)gm * K + gk] : (bf16)0.f;
      }
      for (int idx = tid; idx < 2048; idx += 128) {
        int n = idx >> 5, cc = idx & 31;
        int gk = kt + cc, gn = nBase + n;
        Bt[n][cc] = (gk < K && gn < N) ? (bf16)W[(size_t)gk * N + gn] : (bf16)0.f;
      }
    }
    __syncthreads();
    v16bf af  = frag_b128(&As[0][0], 40, wm, 0, lane);
    v16bf bf0 = frag_b128(&Bt[0][0], 40, wn0, 0, lane);
    v16bf bf1 = frag_b128(&Bt[0][0], 40, wn0 + 32, 0, lane);
    acc0 = WMMA_BF16(af, bf0, acc0);
    acc1 = WMMA_BF16(af, bf1, acc1);
    __syncthreads();
  }

  int mrow = mBase + wm + ((lane >> 4) << 3);
#pragma unroll
  for (int half = 0; half < 2; ++half) {
    int n = nBase + wn0 + half * 32 + (lane & 15);
    const v8f& acc = half ? acc1 : acc0;
#pragma unroll
    for (int r = 0; r < 8; ++r) {
      int m = mrow + r;
      if (m < M && n < N) {
        float v = acc[r];
        if (bias) v += bias[n];
        if (act == 1)       v = 0.5f * v * (1.f + erff(v * 0.70710678118654752f));
        else if (act == 2)  v = 1.f / (1.f + expf(-v));
        else if (act == 3)  v = fmaxf(v, 0.f);
        if (rowscale) v *= rowscale[(m / rpg) * N + n];
        size_t o = (size_t)m * N + n;
        if (resid_mode == 1)      v += resid[o];
        else if (resid_mode == 2) v = resid[o] * v;
        C[o] = v;
      }
    }
  }
}

// ---------------------------------------------------------------------------
// Windowed attention: one block (128 thr, 4 waves) per (window, head).
// N=64 tokens, head_dim=32 (all 4 stages). WMMA for S=qk^T and P@V; b128
// fragment loads (V staged transposed so its fragments are row-contiguous).
// ---------------------------------------------------------------------------
__device__ __forceinline__ int reg3(int a, int R) { return (a < R - 8) ? 0 : ((a < R - 4) ? 1 : 2); }

__global__ void attn_kernel(const float* __restrict__ qkv, const float* __restrict__ rpb,
                            float* __restrict__ out, int Cc, int heads, int nWr, int shifted, int R) {
  int blk = blockIdx.x;
  int w   = blk / heads;
  int h   = blk % heads;
  __shared__ bf16  qs[64][40];    // [token][d]
  __shared__ bf16  ks[64][40];    // [token][d]
  __shared__ bf16  vsT[32][72];   // [d][token]  (transposed V)
  __shared__ float S[64][65];
  __shared__ bf16  P[64][72];
  int tid = threadIdx.x;
  int C3  = 3 * Cc;
  const float* base = qkv + (size_t)w * 64 * C3;
  for (int idx = tid; idx < 64 * 32; idx += 128) {
    int n = idx >> 5, d = idx & 31;
    int col = h * 32 + d;
    qs[n][d]  = (bf16)(base[(size_t)n * C3 + col] * 0.17677669529663687f);  // hd^-0.5, hd=32
    ks[n][d]  = (bf16)(base[(size_t)n * C3 + Cc + col]);
    vsT[d][n] = (bf16)(base[(size_t)n * C3 + 2 * Cc + col]);
  }
  __syncthreads();

  int wave = tid >> 5, lane = tid & 31;
  int m0 = wave * 16;
  int nW = nWr * nWr;
  int widx = w % nW;
  int wr = widx / nWr, wc = widx % nWr;

  // S = (q * scale) @ k^T  + rel-pos bias (+ shift mask)
#pragma unroll
  for (int n0 = 0; n0 < 64; n0 += 16) {
    v8f acc = {0.f, 0.f, 0.f, 0.f, 0.f, 0.f, 0.f, 0.f};
    v16bf af = frag_b128(&qs[0][0], 40, m0, 0, lane);
    v16bf bf = frag_b128(&ks[0][0], 40, n0, 0, lane);  // B[kk][n] = ks[n][kk]
    acc = WMMA_BF16(af, bf, acc);
    int j    = n0 + (lane & 15);
    int mrow = m0 + ((lane >> 4) << 3);
#pragma unroll
    for (int r = 0; r < 8; ++r) {
      int i  = mrow + r;
      int ri = i >> 3, ci = i & 7, rj = j >> 3, cj = j & 7;
      int rel = (ri - rj + 7) * 15 + (ci - cj + 7);
      float v = acc[r] + rpb[rel * heads + h];
      if (shifted) {
        int li = reg3(wr * 8 + ri, R) * 3 + reg3(wc * 8 + ci, R);
        int lj = reg3(wr * 8 + rj, R) * 3 + reg3(wc * 8 + cj, R);
        if (li != lj) v += -100.f;
      }
      S[i][j] = v;
    }
  }
  __syncthreads();

  // softmax rows (fp32), write bf16 P
  if (tid < 64) {
    float mx = -3.4e38f;
    for (int j = 0; j < 64; ++j) mx = fmaxf(mx, S[tid][j]);
    float sum = 0.f;
    for (int j = 0; j < 64; ++j) { float e = expf(S[tid][j] - mx); S[tid][j] = e; sum += e; }
    float inv = 1.f / sum;
    for (int j = 0; j < 64; ++j) P[tid][j] = (bf16)(S[tid][j] * inv);
  }
  __syncthreads();

  // O = P @ V  -> out[w*64+i][h*32 + d]
#pragma unroll
  for (int n0 = 0; n0 < 32; n0 += 16) {
    v8f acc = {0.f, 0.f, 0.f, 0.f, 0.f, 0.f, 0.f, 0.f};
#pragma unroll
    for (int kt = 0; kt < 64; kt += 32) {
      v16bf af = frag_b128(&P[0][0], 72, m0, kt, lane);
      v16bf bf = frag_b128(&vsT[0][0], 72, n0, kt, lane);  // B[kk][n] = vsT[n][kk]
      acc = WMMA_BF16(af, bf, acc);
    }
    int j    = n0 + (lane & 15);
    int mrow = m0 + ((lane >> 4) << 3);
#pragma unroll
    for (int r = 0; r < 8; ++r) {
      int i = mrow + r;
      out[((size_t)w * 64 + i) * Cc + h * 32 + j] = acc[r];
    }
  }
}

// ---------------------------------------------------------------------------
// Small helper kernels
// ---------------------------------------------------------------------------
__global__ void patch_embed_kernel(const float* __restrict__ x, const float* __restrict__ w,
                                   const float* __restrict__ b, float* __restrict__ h) {
  size_t idx = (size_t)blockIdx.x * blockDim.x + threadIdx.x;
  const size_t total = (size_t)4 * 16384 * 128;
  if (idx >= total) return;
  int d = idx & 127;
  size_t t = idx >> 7;
  int w0 = t % 128; t /= 128;
  int h0 = t % 128;
  int bb = (int)(t / 128);
  float acc = b[d];
#pragma unroll
  for (int p = 0; p < 2; ++p)
#pragma unroll
    for (int q = 0; q < 2; ++q)
      for (int c = 0; c < 3; ++c)
        acc += x[(((size_t)bb * 3 + c) * 256 + (2 * h0 + p)) * 256 + (2 * w0 + q)] *
               w[((p * 2 + q) * 3 + c) * 128 + d];
  h[idx] = acc;
}

__global__ void ln_kernel(const float* __restrict__ x, const float* __restrict__ g,
                          const float* __restrict__ b, float* __restrict__ y, int C) {
  int row = blockIdx.x;
  const float* xr = x + (size_t)row * C;
  __shared__ float red[128];
  int tid = threadIdx.x;
  float s = 0.f;
  for (int j = tid; j < C; j += 128) s += xr[j];
  red[tid] = s; __syncthreads();
  for (int st = 64; st > 0; st >>= 1) { if (tid < st) red[tid] += red[tid + st]; __syncthreads(); }
  float mean = red[0] / C;
  __syncthreads();
  float s2 = 0.f;
  for (int j = tid; j < C; j += 128) { float d = xr[j] - mean; s2 += d * d; }
  red[tid] = s2; __syncthreads();
  for (int st = 64; st > 0; st >>= 1) { if (tid < st) red[tid] += red[tid + st]; __syncthreads(); }
  float inv = rsqrtf(red[0] / C + 1e-5f);
  for (int j = tid; j < C; j += 128)
    y[(size_t)row * C + j] = (xr[j] - mean) * inv * g[j] + b[j];
}

// h [4, R, R, C] -> windows xw [4*nW*64, C], with cyclic roll by -shift
__global__ void winpart_kernel(const float* __restrict__ h, float* __restrict__ xw,
                               int R, int C, int shift) {
  size_t idx = (size_t)blockIdx.x * blockDim.x + threadIdx.x;
  size_t total = (size_t)4 * R * R * C;
  if (idx >= total) return;
  int nWr = R >> 3;
  int c = idx % C; size_t t = idx / C;
  int p = t % 64; t /= 64;
  int pc = p & 7, pr = p >> 3;
  int wc = t % nWr; t /= nWr;
  int wr = t % nWr;
  int b  = (int)(t / nWr);
  int sr = (wr * 8 + pr + shift) % R;
  int sc = (wc * 8 + pc + shift) % R;
  xw[idx] = h[(((size_t)b * R + sr) * R + sc) * C + c];
}

// proj (window-major) -> image layout with roll back by +shift, plus residual
__global__ void winunpart_resid_kernel(const float* __restrict__ proj,
                                       const float* __restrict__ shortcut,
                                       float* __restrict__ hout, int R, int C, int shift) {
  size_t idx = (size_t)blockIdx.x * blockDim.x + threadIdx.x;
  size_t total = (size_t)4 * R * R * C;
  if (idx >= total) return;
  int nWr = R >> 3;
  int c = idx % C; size_t t = idx / C;
  int cc = t % R; t /= R;
  int r  = t % R;
  int b  = (int)(t / R);
  int rs = (r + R - shift) % R;
  int cs = (cc + R - shift) % R;
  int wr = rs >> 3, pr = rs & 7, wc = cs >> 3, pc = cs & 7;
  size_t o = ((((size_t)b * nWr + wr) * nWr + wc) * 64 + (pr * 8 + pc)) * C + c;
  hout[idx] = shortcut[idx] + proj[o];
}

// PatchMerging gather: h[4,2R,2R,Ci] -> out[4,R*R,4Ci], concat order 00,10,01,11
__global__ void merge_gather_kernel(const float* __restrict__ h, float* __restrict__ out,
                                    int R, int Ci) {
  size_t idx = (size_t)blockIdx.x * blockDim.x + threadIdx.x;
  int C4 = 4 * Ci;
  size_t total = (size_t)4 * R * R * C4;
  if (idx >= total) return;
  int c4 = idx % C4; size_t t = idx / C4;
  int cc = t % R; t /= R;
  int r  = t % R;
  int b  = (int)(t / R);
  int blkc = c4 / Ci, c = c4 % Ci;
  int dr = (blkc == 1 || blkc == 3) ? 1 : 0;
  int dc = (blkc >= 2) ? 1 : 0;
  int R2 = 2 * R;
  out[idx] = h[(((size_t)b * R2 + (2 * r + dr)) * R2 + (2 * cc + dc)) * Ci + c];
}

// adaptive_mod: per batch: sigmoid(relu(relu(s*w1+b1)@w2+b2)@w3+b3) -> out[b*M..]
__global__ void admod_kernel(const float* __restrict__ s,
                             const float* __restrict__ w1, const float* __restrict__ b1,
                             const float* __restrict__ w2, const float* __restrict__ b2,
                             const float* __restrict__ w3, const float* __restrict__ b3,
                             float* __restrict__ out, int Md) {
  int b = blockIdx.x;
  __shared__ float t1[512];
  __shared__ float t2[512];
  float sv = s[b];
  for (int j = threadIdx.x; j < Md; j += blockDim.x)
    t1[j] = fmaxf(0.f, sv * w1[j] + b1[j]);
  __syncthreads();
  for (int j = threadIdx.x; j < Md; j += blockDim.x) {
    float acc = b2[j];
    for (int k = 0; k < Md; ++k) acc += t1[k] * w2[k * Md + j];
    t2[j] = fmaxf(0.f, acc);
  }
  __syncthreads();
  for (int j = threadIdx.x; j < Md; j += blockDim.x) {
    float acc = b3[j];
    for (int k = 0; k < Md; ++k) acc += t2[k] * w3[k * Md + j];
    out[b * Md + j] = 1.f / (1.f + expf(-acc));
  }
}

// ---------------------------------------------------------------------------
// Host orchestration
// ---------------------------------------------------------------------------
struct BlockP { const float *fc1b,*fc1w,*fc2b,*fc2w,*n1b,*n1g,*n2b,*n2g,*projb,*projw,*qkvb,*qkvw,*rpb; };
struct AdModP { const float *b1,*b2,*b3,*w1,*w2,*w3; };
struct LinP   { const float *b,*w; };

static inline void launch_gemm(hipStream_t st, const float* A, const float* W, const float* bias,
                               const float* resid, int resid_mode, const float* rowscale, int rpg,
                               float* C, int M, int N, int K, int act) {
  dim3 grid((N + 63) / 64, (M + 31) / 32);
  gemm_kernel<<<grid, 128, 0, st>>>(A, W, bias, resid, rowscale, C, M, N, K, act, resid_mode, rpg);
}

extern "C" void kernel_launch(void* const* d_in, const int* in_sizes, int n_in,
                              void* d_out, int out_size, void* d_ws, size_t ws_size,
                              hipStream_t stream) {
  (void)in_sizes; (void)n_in; (void)out_size; (void)ws_size;
  const int EMBv[4]  = {128, 192, 256, 320};
  const int DEP[4]   = {2, 2, 6, 2};
  const int HEADSv[4]= {4, 6, 8, 10};

  const float* x   = (const float*)d_in[0];
  const float* snr = (const float*)d_in[1];
  const float* cr  = (const float*)d_in[2];

  // --- walk flattened params (jax pytree: sorted dict keys) ---
  int pi = 3;
  auto nx = [&]() { return (const float*)d_in[pi++]; };
  AdModP bm[7], bm1[7];
  for (int i = 0; i < 7; ++i) { bm[i].b1=nx(); bm[i].b2=nx(); bm[i].b3=nx(); bm[i].w1=nx(); bm[i].w2=nx(); bm[i].w3=nx(); }
  for (int i = 0; i < 7; ++i) { bm1[i].b1=nx(); bm1[i].b2=nx(); bm1[i].b3=nx(); bm1[i].w1=nx(); bm1[i].w2=nx(); bm1[i].w3=nx(); }
  const float* head_b = nx();
  const float* head_w = nx();
  BlockP blocks[4][6];
  const float *dnb[4] = {nullptr,nullptr,nullptr,nullptr};
  const float *dng[4] = {nullptr,nullptr,nullptr,nullptr};
  const float *drw[4] = {nullptr,nullptr,nullptr,nullptr};
  for (int i = 0; i < 4; ++i) {
    for (int j = 0; j < DEP[i]; ++j) {
      BlockP& B = blocks[i][j];
      B.fc1b=nx(); B.fc1w=nx(); B.fc2b=nx(); B.fc2w=nx();
      B.n1b=nx();  B.n1g=nx();  B.n2b=nx();  B.n2g=nx();
      B.projb=nx(); B.projw=nx(); B.qkvb=nx(); B.qkvw=nx(); B.rpb=nx();
    }
    if (i > 0) { dnb[i]=nx(); dng[i]=nx(); drw[i]=nx(); }
  }
  const float* norm_b = nx();
  const float* norm_g = nx();
  const float* pe_b = nx();
  const float* pe_w = nx();
  LinP sm[8], sm1[8];
  for (int i = 0; i < 8; ++i) { sm[i].b = nx(); sm[i].w = nx(); }
  for (int i = 0; i < 8; ++i) { sm1[i].b = nx(); sm1[i].w = nx(); }

  // --- workspace layout (floats) ---
  float* wsf = (float*)d_ws;
  const size_t MF = (size_t)1 << 20;
  float* HA   = wsf;              // h ping   (<= 8.39M floats)
  float* HB   = wsf + 16 * MF;    // h pong
  float* XW   = wsf + 32 * MF;    // windows / LN-out / attn-out
  float* QKV  = wsf + 48 * MF;    // qkv & fc1 hidden (<= 33.6M floats)
  float* AO   = wsf + 88 * MF;    // proj out
  float* MG   = wsf + 104 * MF;   // LN buf / merge gather / gate out
  float* TA   = wsf + 120 * MF;   // modulation temp ping
  float* TB   = wsf + 121 * MF;   // modulation temp pong
  float* MODV = wsf + 122 * MF;   // adaptive-mod vectors

  // --- patch embed: x[4,3,256,256] -> HA [4, 16384, 128] ---
  {
    size_t total = (size_t)4 * 16384 * 128;
    patch_embed_kernel<<<(int)((total + 255) / 256), 256, 0, stream>>>(x, pe_w, pe_b, HA);
  }
  float* hc = HA;
  float* ho = HB;

  // --- 4 swin stages ---
  for (int i = 0; i < 4; ++i) {
    int R = 128 >> i, C = EMBv[i], heads = HEADSv[i];
    int nWr = R >> 3, nW = nWr * nWr;
    int rows = 4 * R * R;
    size_t elems = (size_t)rows * C;
    int g1 = (int)((elems + 255) / 256);

    if (i > 0) {  // PatchMerging
      int Ci = EMBv[i - 1];
      size_t me = (size_t)rows * 4 * Ci;
      merge_gather_kernel<<<(int)((me + 255) / 256), 256, 0, stream>>>(hc, MG, R, Ci);
      ln_kernel<<<rows, 128, 0, stream>>>(MG, dng[i], dnb[i], XW, 4 * Ci);
      launch_gemm(stream, XW, drw[i], nullptr, nullptr, 0, nullptr, 1, ho, rows, C, 4 * Ci, 0);
      float* t = hc; hc = ho; ho = t;
    }

    for (int j = 0; j < DEP[i]; ++j) {
      const BlockP& B = blocks[i][j];
      int shift = (j % 2 == 0) ? 0 : 4;
      // attention branch
      ln_kernel<<<rows, 128, 0, stream>>>(hc, B.n1g, B.n1b, MG, C);
      winpart_kernel<<<g1, 256, 0, stream>>>(MG, XW, R, C, shift);
      launch_gemm(stream, XW, B.qkvw, B.qkvb, nullptr, 0, nullptr, 1, QKV, rows, 3 * C, C, 0);
      attn_kernel<<<4 * nW * heads, 128, 0, stream>>>(QKV, B.rpb, XW, C, heads, nWr, shift > 0, R);
      launch_gemm(stream, XW, B.projw, B.projb, nullptr, 0, nullptr, 1, AO, rows, C, C, 0);
      winunpart_resid_kernel<<<g1, 256, 0, stream>>>(AO, hc, ho, R, C, shift);
      { float* t = hc; hc = ho; ho = t; }
      // MLP branch
      ln_kernel<<<rows, 128, 0, stream>>>(hc, B.n2g, B.n2b, MG, C);
      launch_gemm(stream, MG, B.fc1w, B.fc1b, nullptr, 0, nullptr, 1, QKV, rows, 4 * C, C, 1);
      launch_gemm(stream, QKV, B.fc2w, B.fc2b, hc, 1, nullptr, 1, ho, rows, C, 4 * C, 0);
      { float* t = hc; hc = ho; ho = t; }
    }
  }

  // --- final norm: hc [4,256,320] -> XW ---
  ln_kernel<<<1024, 128, 0, stream>>>(hc, norm_g, norm_b, XW, 320);
  float* hn = XW;

  // --- SNR-adaptive modulation (HID = 480) ---
  for (int i = 0; i < 7; ++i)
    admod_kernel<<<4, 128, 0, stream>>>(snr, bm[i].w1, bm[i].b1, bm[i].w2, bm[i].b2,
                                        bm[i].w3, bm[i].b3, MODV + i * 2048, 480);
  launch_gemm(stream, hn, sm[0].w, sm[0].b, nullptr, 0, MODV + 0, 256, TA, 1024, 480, 320, 0);
  {
    float* ta = TA; float* tb = TB;
    for (int i = 1; i < 7; ++i) {
      launch_gemm(stream, ta, sm[i].w, sm[i].b, nullptr, 0, MODV + i * 2048, 256, tb, 1024, 480, 480, 0);
      float* t = ta; ta = tb; tb = t;
    }
    // gate: h = hn * sigmoid(temp @ sm[7].w + b)  -> MG [1024,320]
    launch_gemm(stream, ta, sm[7].w, sm[7].b, hn, 2, nullptr, 1, MG, 1024, 320, 480, 2);
  }
  // head: [1024,320] @ [320,32] -> HA [1024,32]
  launch_gemm(stream, MG, head_w, head_b, nullptr, 0, nullptr, 1, HA, 1024, 32, 320, 0);

  // --- CR-adaptive modulation (HID_RA = 48) ---
  float* MODV1 = MODV + 7 * 2048;
  for (int i = 0; i < 7; ++i)
    admod_kernel<<<4, 128, 0, stream>>>(cr, bm1[i].w1, bm1[i].b1, bm1[i].w2, bm1[i].b2,
                                        bm1[i].w3, bm1[i].b3, MODV1 + i * 256, 48);
  launch_gemm(stream, HA, sm1[0].w, sm1[0].b, nullptr, 0, MODV1 + 0, 256, TA, 1024, 48, 32, 0);
  {
    float* ta = TA; float* tb = TB;
    for (int i = 1; i < 7; ++i) {
      launch_gemm(stream, ta, sm1[i].w, sm1[i].b, nullptr, 0, MODV1 + i * 256, 256, tb, 1024, 48, 48, 0);
      float* t = ta; ta = tb; tb = t;
    }
    // final gate -> d_out [4,256,32] fp32
    launch_gemm(stream, ta, sm1[7].w, sm1[7].b, HA, 2, nullptr, 1, (float*)d_out, 1024, 32, 48, 2);
  }
}
